// EdgeGatedGraphConv_69131793596856
// MI455X (gfx1250) — matrix operand order repaired and checked by reading, not compiled, not run on hardware
//
#include <hip/hip_runtime.h>

#define D 128
#define LN_EPS 1e-5f
#define GATE_EPS 1e-6f

typedef __attribute__((ext_vector_type(16))) _Float16 v16h;
typedef __attribute__((ext_vector_type(8)))  float    v8f;

// ---------------------------------------------------------------------------
// Helpers
// ---------------------------------------------------------------------------
__device__ inline float sigm(float z) { return 1.f / (1.f + __expf(-z)); }
__device__ inline float silu(float z) { return z / (1.f + __expf(-z)); }

// Load one 16x32 f16 A-fragment for this lane from a row-major f32 matrix.
// lane 0-15: M=lane,   K = kb..kb+7 and kb+16..kb+23 with kb = 32*c + 0
// lane16-31: M=lane-16, same but kb = 32*c + 8
__device__ inline v16h load_a_frag(const float* __restrict__ row, int kb) {
    float4 f0 = *(const float4*)(row + kb);
    float4 f1 = *(const float4*)(row + kb + 4);
    float4 f2 = *(const float4*)(row + kb + 16);
    float4 f3 = *(const float4*)(row + kb + 20);
    v16h a;
    a[0]  = (_Float16)f0.x; a[1]  = (_Float16)f0.y; a[2]  = (_Float16)f0.z; a[3]  = (_Float16)f0.w;
    a[4]  = (_Float16)f1.x; a[5]  = (_Float16)f1.y; a[6]  = (_Float16)f1.z; a[7]  = (_Float16)f1.w;
    a[8]  = (_Float16)f2.x; a[9]  = (_Float16)f2.y; a[10] = (_Float16)f2.z; a[11] = (_Float16)f2.w;
    a[12] = (_Float16)f3.x; a[13] = (_Float16)f3.y; a[14] = (_Float16)f3.z; a[15] = (_Float16)f3.w;
    return a;
}

// ---------------------------------------------------------------------------
// K_w: pre-convert the 5 weight matrices (f32 row-major [K=128][N=128]) into
// f16 WMMA B-fragment layout:
//   frag index = ((mat*8 + ntile)*4 + kchunk)*32*16 + lane*16 + e
//   lane 0-15 : N = 16*ntile + lane,     K = 32*kchunk + (e&7) + (e>>3)*16      (e = 0..15)
//   lane16-31 : N = 16*ntile + lane-16,  K = 32*kchunk + 8 + (e&7) + (e>>3)*16
// ---------------------------------------------------------------------------
__global__ void wconv_kernel(const float* __restrict__ W0, const float* __restrict__ W1,
                             const float* __restrict__ W2, const float* __restrict__ W3,
                             const float* __restrict__ W4, _Float16* __restrict__ wf16) {
    int tid = blockIdx.x * blockDim.x + threadIdx.x;
    if (tid >= 5 * 16384) return;
    int mat  = tid >> 14;
    int idx  = tid & 16383;
    int e    = idx & 15;
    int lane = (idx >> 4) & 31;
    int c    = (idx >> 9) & 3;
    int jj   = (idx >> 11) & 7;
    int n = 16 * jj + (lane & 15);
    int k = 32 * c + ((lane >> 4) & 1) * 8 + (e & 7) + (e >> 3) * 16;
    const float* W = (mat == 0) ? W0 : (mat == 1) ? W1 : (mat == 2) ? W2 : (mat == 3) ? W3 : W4;
    wf16[tid] = (_Float16)W[k * D + n];
}

// ---------------------------------------------------------------------------
// K0: zero num/den accumulators (must happen every call: graph replays)
// ---------------------------------------------------------------------------
__global__ void zero_kernel(float* __restrict__ p, size_t n4) {
    size_t i = (size_t)blockIdx.x * blockDim.x + threadIdx.x;
    size_t stride = (size_t)gridDim.x * blockDim.x;
    float4 z = {0.f, 0.f, 0.f, 0.f};
    for (size_t idx = i; idx < n4; idx += stride) ((float4*)p)[idx] = z;
}

// ---------------------------------------------------------------------------
// K1: node-side GEMMs. One wave per TWO 16-node tiles (32 rows): each loaded
// B fragment feeds two back-to-back WMMAs (halves B traffic + wait stalls).
// __launch_bounds__(128,1): relax VGPR budget so nothing spills to scratch.
//   e_src = nf@Wsg+b, e_dst = nf@Wdg+b, Ax = nf@Wsu+b (-> d_out x region),
//   Bh = nf@Wdu+b.   4 mats x 8 ntiles x 4 chunks x 2 Mtiles = 256 wmma/wave.
// ---------------------------------------------------------------------------
__global__ void __launch_bounds__(128, 1)
node_gemm_kernel(const float* __restrict__ nf, const _Float16* __restrict__ wf16,
                 const float* __restrict__ b0, const float* __restrict__ b1,
                 const float* __restrict__ b3, const float* __restrict__ b4,
                 float* __restrict__ e_src, float* __restrict__ e_dst,
                 float* __restrict__ Ax, float* __restrict__ Bh, int n_tiles) {
    int wv = blockIdx.x * (blockDim.x >> 5) + (threadIdx.x >> 5);
    int t0 = wv * 2;
    if (t0 >= n_tiles) return;
    int t1 = t0 + 1;
    bool has2 = (t1 < n_tiles);
    int t1c = has2 ? t1 : t0;  // clamped (loads only; stores guarded)
    int lane = threadIdx.x & 31;
    int mlo = lane & 15;
    int hi = (lane >> 4) & 1;

    const float* row0 = nf + (size_t)(t0 * 16 + mlo) * D;
    const float* row1 = nf + (size_t)(t1c * 16 + mlo) * D;
    v16h a0[4], a1[4];
#pragma unroll
    for (int c = 0; c < 4; ++c) {
        a0[c] = load_a_frag(row0, 32 * c + hi * 8);
        a1[c] = load_a_frag(row1, 32 * c + hi * 8);
    }

    const int mats[4] = {0, 1, 3, 4};
    const float* biases[4] = {b0, b1, b3, b4};
    float* outs[4] = {e_src, e_dst, Ax, Bh};

#pragma unroll
    for (int w = 0; w < 4; ++w) {
        const _Float16* wbase = wf16 + (size_t)mats[w] * 16384;
        float bj[8];  // hoisted bias: one value per ntile j
#pragma unroll
        for (int j = 0; j < 8; ++j) bj[j] = biases[w][16 * j + mlo];
#pragma unroll
        for (int j = 0; j < 8; ++j) {
            v8f acc0, acc1;
#pragma unroll
            for (int r = 0; r < 8; ++r) { acc0[r] = bj[j]; acc1[r] = bj[j]; }
#pragma unroll
            for (int c = 0; c < 4; ++c) {
                v16h bf = *(const v16h*)(wbase + (((size_t)j * 4 + c) * 32 + lane) * 16);
                acc0 = __builtin_amdgcn_wmma_f32_16x16x32_f16(false, a0[c], false, bf,
                                                              (short)0, acc0, false, false);
                acc1 = __builtin_amdgcn_wmma_f32_16x16x32_f16(false, a1[c], false, bf,
                                                              (short)0, acc1, false, false);
            }
            float* op0 = outs[w] + (size_t)(t0 * 16 + hi * 8) * D + 16 * j + mlo;
#pragma unroll
            for (int r = 0; r < 8; ++r) op0[(size_t)r * D] = acc0[r];
            if (has2) {
                float* op1 = outs[w] + (size_t)(t1 * 16 + hi * 8) * D + 16 * j + mlo;
#pragma unroll
                for (int r = 0; r < 8; ++r) op1[(size_t)r * D] = acc1[r];
            }
        }
    }
}

// ---------------------------------------------------------------------------
// K2: edge-side GEMM + gather + sigmoid + scatter.  One wave per TWO 16-edge
// tiles. __launch_bounds__(128,1) to avoid VGPR spills (this kernel carries
// 2x A fragments + 32 edge indices + gather/atomic temporaries).
//   m = ef@Weg + b + e_src[src] + e_dst[dst]   -> staged into d_out y region
//   sigma = sigmoid(m); atomically accumulate num += sigma*Bh[src], den += sigma
//   (gather tables + num/den are L2-resident: 5 x 25.6 MB << 192 MB L2)
// ---------------------------------------------------------------------------
__global__ void __launch_bounds__(128, 1)
edge_gemm_kernel(const float* __restrict__ ef, const _Float16* __restrict__ wf16,
                 const float* __restrict__ b_edge,
                 const float* __restrict__ e_src, const float* __restrict__ e_dst,
                 const float* __restrict__ Bh,
                 const int* __restrict__ src, const int* __restrict__ dst,
                 float* __restrict__ m_out, float* __restrict__ num,
                 float* __restrict__ den, int n_tiles) {
    int wv = blockIdx.x * (blockDim.x >> 5) + (threadIdx.x >> 5);
    int t0 = wv * 2;
    if (t0 >= n_tiles) return;
    int t1 = t0 + 1;
    bool has2 = (t1 < n_tiles);
    int t1c = has2 ? t1 : t0;
    int lane = threadIdx.x & 31;
    int mlo = lane & 15;
    int hi = (lane >> 4) & 1;

    const float* row0 = ef + (size_t)(t0 * 16 + mlo) * D;
    const float* row1 = ef + (size_t)(t1c * 16 + mlo) * D;
    v16h a0[4], a1[4];
#pragma unroll
    for (int c = 0; c < 4; ++c) {
        a0[c] = load_a_frag(row0, 32 * c + hi * 8);
        a1[c] = load_a_frag(row1, 32 * c + hi * 8);
    }

    int eb0 = t0 * 16 + hi * 8;
    int eb1 = t1c * 16 + hi * 8;
    int se0[8], de0[8], se1[8], de1[8];
#pragma unroll
    for (int r = 0; r < 8; ++r) {
        se0[r] = src[eb0 + r]; de0[r] = dst[eb0 + r];
        se1[r] = src[eb1 + r]; de1[r] = dst[eb1 + r];
    }

    const _Float16* wbase = wf16 + (size_t)2 * 16384;  // W_edge_gate
    float bj[8];
#pragma unroll
    for (int j = 0; j < 8; ++j) bj[j] = b_edge[16 * j + mlo];

#pragma unroll
    for (int j = 0; j < 8; ++j) {
        v8f acc0, acc1;
#pragma unroll
        for (int r = 0; r < 8; ++r) { acc0[r] = bj[j]; acc1[r] = bj[j]; }
#pragma unroll
        for (int c = 0; c < 4; ++c) {
            v16h bf = *(const v16h*)(wbase + (((size_t)j * 4 + c) * 32 + lane) * 16);
            acc0 = __builtin_amdgcn_wmma_f32_16x16x32_f16(false, a0[c], false, bf,
                                                          (short)0, acc0, false, false);
            acc1 = __builtin_amdgcn_wmma_f32_16x16x32_f16(false, a1[c], false, bf,
                                                          (short)0, acc1, false, false);
        }
        int n = 16 * j + mlo;
#pragma unroll
        for (int r = 0; r < 8; ++r) {
            size_t so = (size_t)se0[r] * D + n;
            size_t dofs = (size_t)de0[r] * D + n;
            float mv = acc0[r] + e_src[so] + e_dst[dofs];
            m_out[(size_t)(eb0 + r) * D + n] = mv;
            float sg = sigm(mv);
            unsafeAtomicAdd(&den[dofs], sg);
            unsafeAtomicAdd(&num[dofs], sg * Bh[so]);
        }
        if (has2) {
#pragma unroll
            for (int r = 0; r < 8; ++r) {
                size_t so = (size_t)se1[r] * D + n;
                size_t dofs = (size_t)de1[r] * D + n;
                float mv = acc1[r] + e_src[so] + e_dst[dofs];
                m_out[(size_t)(eb1 + r) * D + n] = mv;
                float sg = sigm(mv);
                unsafeAtomicAdd(&den[dofs], sg);
                unsafeAtomicAdd(&num[dofs], sg * Bh[so]);
            }
        }
    }
}

// ---------------------------------------------------------------------------
// K3: node epilogue. One wave per row (4 floats/lane):
//   x = Ax + num/(den+eps); x = nf + silu(layernorm(x));  in-place in d_out x.
// ---------------------------------------------------------------------------
__global__ void node_final_kernel(const float* __restrict__ nf, const float* __restrict__ num,
                                  const float* __restrict__ den, const float* __restrict__ lnw,
                                  const float* __restrict__ lnb, float* __restrict__ x_out,
                                  int n_rows) {
    int row = blockIdx.x * (blockDim.x >> 5) + (threadIdx.x >> 5);
    if (row >= n_rows) return;
    int lane = threadIdx.x & 31;
    size_t base = (size_t)row * D + lane * 4;

    float4 ax = *(const float4*)(x_out + base);
    float4 nu = *(const float4*)(num + base);
    float4 de = *(const float4*)(den + base);
    float x0 = ax.x + nu.x / (de.x + GATE_EPS);
    float x1 = ax.y + nu.y / (de.y + GATE_EPS);
    float x2 = ax.z + nu.z / (de.z + GATE_EPS);
    float x3 = ax.w + nu.w / (de.w + GATE_EPS);

    float sum = x0 + x1 + x2 + x3;
    float sq = x0 * x0 + x1 * x1 + x2 * x2 + x3 * x3;
#pragma unroll
    for (int o = 16; o > 0; o >>= 1) {
        sum += __shfl_xor(sum, o, 32);
        sq  += __shfl_xor(sq, o, 32);
    }
    float mu = sum * (1.f / (float)D);
    float var = sq * (1.f / (float)D) - mu * mu;
    float rs = rsqrtf(var + LN_EPS);

    float4 w = *(const float4*)(lnw + lane * 4);
    float4 b = *(const float4*)(lnb + lane * 4);
    float4 res = *(const float4*)(nf + base);
    float4 out;
    out.x = res.x + silu((x0 - mu) * rs * w.x + b.x);
    out.y = res.y + silu((x1 - mu) * rs * w.y + b.y);
    out.z = res.z + silu((x2 - mu) * rs * w.z + b.z);
    out.w = res.w + silu((x3 - mu) * rs * w.w + b.w);
    *(float4*)(x_out + base) = out;
}

// ---------------------------------------------------------------------------
// K4: edge epilogue. y = ef + silu(layernorm(m));  m lives in d_out y region.
// ---------------------------------------------------------------------------
__global__ void edge_final_kernel(const float* __restrict__ ef, const float* __restrict__ lnw,
                                  const float* __restrict__ lnb, float* __restrict__ y_out,
                                  int n_rows) {
    int row = blockIdx.x * (blockDim.x >> 5) + (threadIdx.x >> 5);
    if (row >= n_rows) return;
    int lane = threadIdx.x & 31;
    size_t base = (size_t)row * D + lane * 4;

    float4 m = *(const float4*)(y_out + base);
    float sum = m.x + m.y + m.z + m.w;
    float sq = m.x * m.x + m.y * m.y + m.z * m.z + m.w * m.w;
#pragma unroll
    for (int o = 16; o > 0; o >>= 1) {
        sum += __shfl_xor(sum, o, 32);
        sq  += __shfl_xor(sq, o, 32);
    }
    float mu = sum * (1.f / (float)D);
    float var = sq * (1.f / (float)D) - mu * mu;
    float rs = rsqrtf(var + LN_EPS);

    float4 w = *(const float4*)(lnw + lane * 4);
    float4 b = *(const float4*)(lnb + lane * 4);
    float4 res = *(const float4*)(ef + base);
    float4 out;
    out.x = res.x + silu((m.x - mu) * rs * w.x + b.x);
    out.y = res.y + silu((m.y - mu) * rs * w.y + b.y);
    out.z = res.z + silu((m.z - mu) * rs * w.z + b.z);
    out.w = res.w + silu((m.w - mu) * rs * w.w + b.w);
    *(float4*)(y_out + base) = out;
}

// ---------------------------------------------------------------------------
extern "C" void kernel_launch(void* const* d_in, const int* in_sizes, int n_in,
                              void* d_out, int out_size, void* d_ws, size_t ws_size,
                              hipStream_t stream) {
    const float* nf           = (const float*)d_in[0];
    const float* ef           = (const float*)d_in[1];
    const float* W_src_gate   = (const float*)d_in[2];
    const float* b_src_gate   = (const float*)d_in[3];
    const float* W_dst_gate   = (const float*)d_in[4];
    const float* b_dst_gate   = (const float*)d_in[5];
    const float* W_edge_gate  = (const float*)d_in[6];
    const float* b_edge_gate  = (const float*)d_in[7];
    const float* W_src_update = (const float*)d_in[8];
    const float* b_src_update = (const float*)d_in[9];
    const float* W_dst_update = (const float*)d_in[10];
    const float* b_dst_update = (const float*)d_in[11];
    const float* ln_nodes_w   = (const float*)d_in[12];
    const float* ln_nodes_b   = (const float*)d_in[13];
    const float* ln_edges_w   = (const float*)d_in[14];
    const float* ln_edges_b   = (const float*)d_in[15];
    const int*   src          = (const int*)d_in[16];
    const int*   dst          = (const int*)d_in[17];

    const int N = in_sizes[0] / D;   // 50000
    const int E = in_sizes[1] / D;   // 600000

    float* x_out = (float*)d_out;             // [N, D] — also stages Ax
    float* y_out = x_out + (size_t)N * D;     // [E, D] — also stages m

    // Workspace layout (~122 MB): wf16 | e_src | e_dst | Bh | num | den
    char* ws = (char*)d_ws;
    _Float16* wf16 = (_Float16*)ws;
    const size_t NB = (size_t)N * D;
    float* e_src = (float*)(ws + 5 * 16384 * sizeof(_Float16) + 64);
    e_src = (float*)(((uintptr_t)e_src + 255) & ~(uintptr_t)255);
    float* e_dst = e_src + NB;
    float* Bh    = e_dst + NB;
    float* num   = Bh + NB;
    float* den   = num + NB;

    // 1) weights -> f16 B-fragment layout
    wconv_kernel<<<(5 * 16384 + 255) / 256, 256, 0, stream>>>(
        W_src_gate, W_dst_gate, W_edge_gate, W_src_update, W_dst_update, wf16);

    // 2) zero num+den (contiguous 2*NB floats)
    zero_kernel<<<4096, 256, 0, stream>>>(num, (2 * NB) / 4);

    // 3) node GEMMs (WMMA, 2 M-tiles per wave)
    int ntiles = N / 16;
    int nwaves = (ntiles + 1) / 2;
    node_gemm_kernel<<<(nwaves + 3) / 4, 128, 0, stream>>>(
        nf, wf16, b_src_gate, b_dst_gate, b_src_update, b_dst_update,
        e_src, e_dst, x_out, Bh, ntiles);

    // 4) edge GEMM + gather + sigmoid + scatter atomics (WMMA, 2 M-tiles/wave)
    int etiles = E / 16;
    int ewaves = (etiles + 1) / 2;
    edge_gemm_kernel<<<(ewaves + 3) / 4, 128, 0, stream>>>(
        ef, wf16, b_edge_gate, e_src, e_dst, Bh, src, dst, y_out, num, den, etiles);

    // 5) node epilogue: h, layernorm, silu, residual
    node_final_kernel<<<(N + 7) / 8, 256, 0, stream>>>(
        nf, num, den, ln_nodes_w, ln_nodes_b, x_out, N);

    // 6) edge epilogue: layernorm(m), silu, residual
    edge_final_kernel<<<(E + 7) / 8, 256, 0, stream>>>(
        ef, ln_edges_w, ln_edges_b, y_out, E);
}
